// DPMMatchingHead_61503931679104
// MI455X (gfx1250) — matrix operand order repaired and checked by reading, compiled, not run on hardware
//
#include <hip/hip_runtime.h>
#include <math.h>

// ---------------- problem constants (match reference) ----------------
#define B_  32
#define P_  10
#define PP  16          // padded prototype rows for 16x16 WMMA tiles
#define N_  4096
#define D_  768
#define TEMP_ 0.1f
#define TAU0_ 0.1f
#define EPS_  1e-8f

typedef unsigned short bf16_t;
typedef __attribute__((ext_vector_type(16))) __bf16 v16bf;
typedef __attribute__((ext_vector_type(8)))  float  v8f;
typedef __attribute__((ext_vector_type(2)))  float  v2f;

union ABfrag { v16bf v; uint4 q[2]; };

__device__ inline bf16_t f2bf(float f) {          // round-to-nearest-even
    unsigned u = __float_as_uint(f);
    u += 0x7FFFu + ((u >> 16) & 1u);
    return (bf16_t)(u >> 16);
}

// ---------------- tau init ----------------
__global__ void k_init_tau(float* tau) {
    tau[threadIdx.x] = TAU0_;                     // 512 entries = B*PP
}

// ---------------- row L2-normalize fp32 -> bf16 (D=768, 256 thr/row) --------
__global__ void k_norm_rows(const float* __restrict__ src, bf16_t* __restrict__ dst) {
    const size_t row = blockIdx.x;
    const float* r = src + row * D_;
    bf16_t*      o = dst + row * D_;
    const int tid = threadIdx.x;
    float v0 = r[tid], v1 = r[tid + 256], v2 = r[tid + 512];
    __shared__ float red[256];
    red[tid] = v0 * v0 + v1 * v1 + v2 * v2;
    __syncthreads();
    for (int s = 128; s > 0; s >>= 1) { if (tid < s) red[tid] += red[tid + s]; __syncthreads(); }
    const float sc = 1.0f / fmaxf(sqrtf(red[0]), EPS_);
    o[tid] = f2bf(v0 * sc); o[tid + 256] = f2bf(v1 * sc); o[tid + 512] = f2bf(v2 * sc);
}

// ---------------- prototype normalize fp32 -> bf16 padded [B,16,D] ----------
__global__ void k_norm_protos(const float* __restrict__ src, int rowsPerB,
                              bf16_t* __restrict__ protN) {
    const int p = blockIdx.x & 15, b = blockIdx.x >> 4;
    const int tid = threadIdx.x;
    bf16_t* o = protN + ((size_t)b * PP + p) * D_;
    if (p >= P_) { o[tid] = 0; o[tid + 256] = 0; o[tid + 512] = 0; return; }
    const float* r = src + ((size_t)b * rowsPerB + p) * D_;
    float v0 = r[tid], v1 = r[tid + 256], v2 = r[tid + 512];
    __shared__ float red[256];
    red[tid] = v0 * v0 + v1 * v1 + v2 * v2;
    __syncthreads();
    for (int s = 128; s > 0; s >>= 1) { if (tid < s) red[tid] += red[tid + s]; __syncthreads(); }
    const float sc = 1.0f / fmaxf(sqrtf(red[0]), EPS_);
    o[tid] = f2bf(v0 * sc); o[tid + 256] = f2bf(v1 * sc); o[tid + 512] = f2bf(v2 * sc);
}

// ---------------- sim GEMM: S[b,16,n] = protN[b] (16xD) * featsN^T ----------
// bf16 inputs, fp32 accum via v_wmma_f32_16x16x32_bf16. 8 waves/WG, 128 n/WG.
__global__ void k_sim(const bf16_t* __restrict__ protN,
                      const bf16_t* __restrict__ featsN,
                      float* __restrict__ out,
                      int featsPerB,       // 1: featsN is [B,N,D]; 0: shared [N,D]
                      int finalMode) {     // 1: write [(b*10+p)*N + n], p<10 only
    __shared__ bf16_t protLds[PP * D_];    // 24 KB
    const int b = blockIdx.y, tid = threadIdx.x;
    const int nBase = blockIdx.x * 128;

    { // cooperative load of the 16xD prototype tile
        const uint32_t* s = (const uint32_t*)(protN + (size_t)b * PP * D_);
        uint32_t* d = (uint32_t*)protLds;
        for (int i = tid; i < PP * D_ / 2; i += 256) d[i] = s[i];
    }
    __syncthreads();

    const int wave = tid >> 5, lane = tid & 31;
    const int l16 = lane & 15, h = lane >> 4;
    const int nTile = nBase + wave * 16;
    const bf16_t* fRow = featsN + (featsPerB ? (size_t)b * N_ * D_ : (size_t)0)
                                + (size_t)(nTile + l16) * D_;
    v8f acc = {};
#pragma unroll 4
    for (int k0 = 0; k0 < D_; k0 += 32) {
        ABfrag a, bm;
        // A 16x32 bf16: lanes 0-15 K={k0..k0+7, k0+16..23}, lanes 16-31 shift by 8
        a.q[0] = *(const uint4*)(protLds + (size_t)l16 * D_ + k0 + h * 8);
        a.q[1] = *(const uint4*)(protLds + (size_t)l16 * D_ + k0 + 16 + h * 8);
        // B 32x16 bf16: column n=l16, lanes 0-15 K={k0..k0+7, k0+16..23}, lanes 16-31 shift by 8
        bm.q[0] = *(const uint4*)(fRow + k0 + h * 8);
        bm.q[1] = *(const uint4*)(fRow + k0 + 16 + h * 8);
        acc = __builtin_amdgcn_wmma_f32_16x16x32_bf16(false, a.v, false, bm.v,
                                                      (short)0, acc, false, false);
    }
    if (!finalMode) {
        float* o = out + (size_t)b * PP * N_;
#pragma unroll
        for (int r = 0; r < 8; ++r) o[(size_t)(r + 8 * h) * N_ + nTile + l16] = acc[r];
    } else {
#pragma unroll
        for (int r = 0; r < 8; ++r) {
            const int p = r + 8 * h;
            if (p < P_) out[((size_t)b * P_ + p) * N_ + nTile + l16] = acc[r];
        }
    }
}

// ---------------- softmax over N per (b,p): W = softmax(S/(TEMP*tau)) -------
__global__ void k_softmax(const float* __restrict__ S, const float* __restrict__ tau,
                          float* __restrict__ W) {
    const int p = blockIdx.x, b = blockIdx.y, tid = threadIdx.x;
    const float* row = S + ((size_t)b * PP + p) * N_;
    float*      wrow = W + ((size_t)b * PP + p) * N_;
    const float scale = 1.0f / (TEMP_ * tau[b * PP + p]);
    float loc[16];
    float m = -INFINITY;
#pragma unroll
    for (int i = 0; i < 16; ++i) { loc[i] = row[tid + i * 256] * scale; m = fmaxf(m, loc[i]); }
    __shared__ float red[256];
    red[tid] = m; __syncthreads();
    for (int s = 128; s > 0; s >>= 1) { if (tid < s) red[tid] = fmaxf(red[tid], red[tid + s]); __syncthreads(); }
    m = red[0]; __syncthreads();
    float ssum = 0.f;
#pragma unroll
    for (int i = 0; i < 16; ++i) { loc[i] = __expf(loc[i] - m); ssum += loc[i]; }
    red[tid] = ssum; __syncthreads();
    for (int s = 128; s > 0; s >>= 1) { if (tid < s) red[tid] += red[tid + s]; __syncthreads(); }
    const float inv = 1.0f / red[0];
#pragma unroll
    for (int i = 0; i < 16; ++i) wrow[tid + i * 256] = loc[i] * inv;
}

// ---------------- hard assignment: idx[b,n] = argmax_p W[b,p,n] -------------
__global__ void k_argmax(const float* __restrict__ W, int* __restrict__ idx) {
    const size_t g = (size_t)blockIdx.x * 256 + threadIdx.x;   // < B*N
    const int b = (int)(g >> 12), n = (int)(g & (N_ - 1));
    const float* col = W + (size_t)b * PP * N_ + n;
    float best = col[0]; int bi = 0;
#pragma unroll
    for (int p = 1; p < P_; ++p) {
        const float v = col[(size_t)p * N_];
        if (v > best) { best = v; bi = p; }                    // first-max like jnp.argmax
    }
    idx[g] = bi;
}

// ---------------- update GEMM: proto[b,16,D] = maskedW (16xN) * feats (NxD) --
// fp32 WMMA (16x16x4), mask applied while staging A. 8 waves/WG, 128 d/WG.
#define KT 32
__global__ void k_update(const float* __restrict__ W, const int* __restrict__ idx,
                         const float* __restrict__ feats, float* __restrict__ protoBuf) {
    __shared__ float aLds[PP * KT];        // 2 KB
    __shared__ float bLds[KT * 128];       // 16 KB
    const int b = blockIdx.y, tid = threadIdx.x;
    const int dBase = blockIdx.x * 128;
    const int wave = tid >> 5, lane = tid & 31;
    const int l16 = lane & 15, h = lane >> 4;
    const float* fb = feats + (size_t)b * N_ * D_;
    const float* Wb = W + (size_t)b * PP * N_;
    const int*   ib = idx + (size_t)b * N_;
    v8f acc = {};
#pragma unroll 1
    for (int n0 = 0; n0 < N_; n0 += KT) {
        // stage masked A tile [16 x KT]
        for (int i = tid; i < PP * KT; i += 256) {
            const int p = i / KT, kk = i % KT, n = n0 + kk;
            aLds[i] = (ib[n] == p) ? Wb[(size_t)p * N_ + n] : 0.0f;
        }
        // stage B tile [KT x 128] (coalesced along d)
        for (int i = tid; i < KT * 128; i += 256) {
            const int nn = i >> 7, dd = i & 127;
            bLds[i] = fb[(size_t)(n0 + nn) * D_ + dBase + dd];
        }
        __syncthreads();
#pragma unroll
        for (int kk = 0; kk < KT; kk += 4) {
            v2f av, bv;
            av.x = aLds[l16 * KT + kk + 2 * h];
            av.y = aLds[l16 * KT + kk + 2 * h + 1];
            bv.x = bLds[(kk + 2 * h) * 128 + wave * 16 + l16];
            bv.y = bLds[(kk + 2 * h + 1) * 128 + wave * 16 + l16];
            acc = __builtin_amdgcn_wmma_f32_16x16x4_f32(false, av, false, bv,
                                                        (short)0, acc, false, false);
        }
        __syncthreads();
    }
#pragma unroll
    for (int r = 0; r < 8; ++r) {
        const int p = r + 8 * h;
        if (p < P_) protoBuf[((size_t)b * PP + p) * D_ + dBase + wave * 16 + l16] = acc[r];
    }
}

// ---------------- density -> tau update per (b,p) ---------------------------
__global__ void k_density(const float* __restrict__ S2, const int* __restrict__ idx,
                          float* __restrict__ tau) {
    const int p = blockIdx.x, b = blockIdx.y, tid = threadIdx.x;
    const float* row = S2 + ((size_t)b * PP + p) * N_;
    const int*   ib  = idx + (size_t)b * N_;
    float s = 0.f, c = 0.f;
    for (int n = tid; n < N_; n += 256)
        if (ib[n] == p) { s += row[n]; c += 1.0f; }
    __shared__ float rs[256], rc[256];
    rs[tid] = s; rc[tid] = c; __syncthreads();
    for (int st = 128; st > 0; st >>= 1) {
        if (tid < st) { rs[tid] += rs[tid + st]; rc[tid] += rc[tid + st]; }
        __syncthreads();
    }
    if (tid == 0) {
        const float msum = rc[0];
        const float dens = (msum >= 1.0f) ? rs[0] / fmaxf(msum, 1.0f) : 0.0f;
        tau[b * PP + p] = fmaxf(1.0f - dens, 1e-10f);
    }
}

// ---------------- final prototype copy to output ----------------------------
__global__ void k_copy_protos(const float* __restrict__ protoBuf, float* __restrict__ out) {
    const int g = blockIdx.x * 256 + threadIdx.x;   // < B*P*D
    const int d = g % D_, bp = g / D_, b = bp / P_, p = bp % P_;
    out[g] = protoBuf[((size_t)b * PP + p) * D_ + d];
}

// ======================= host orchestration =================================
extern "C" void kernel_launch(void* const* d_in, const int* in_sizes, int n_in,
                              void* d_out, int out_size, void* d_ws, size_t ws_size,
                              hipStream_t stream) {
    (void)in_sizes; (void)n_in; (void)out_size; (void)ws_size;
    const float* protos0  = (const float*)d_in[0];   // [B,P,D]
    const float* feats    = (const float*)d_in[1];   // [B,N,D]
    const float* featsOrg = (const float*)d_in[2];   // [N,D]

    char* ws = (char*)d_ws;
    size_t off = 0;
    auto alloc = [&](size_t bytes) { void* p = ws + off; off = (off + bytes + 255) & ~(size_t)255; return p; };
    bf16_t* featsN  = (bf16_t*)alloc((size_t)B_ * N_ * D_ * 2);   // normalized feats bf16
    bf16_t* forgN   = (bf16_t*)alloc((size_t)N_ * D_ * 2);        // normalized feats_org bf16
    bf16_t* protN   = (bf16_t*)alloc((size_t)B_ * PP * D_ * 2);   // normalized protos bf16
    float*  S       = (float*) alloc((size_t)B_ * PP * N_ * 4);   // sim buffer
    float*  Wbuf    = (float*) alloc((size_t)B_ * PP * N_ * 4);   // softmax weights
    int*    idx     = (int*)   alloc((size_t)B_ * N_ * 4);        // hard assignment
    float*  protoBuf= (float*) alloc((size_t)B_ * PP * D_ * 4);   // updated prototypes fp32
    float*  tau     = (float*) alloc((size_t)B_ * PP * 4);

    float* outProt = (float*)d_out;                       // [B*P, D]
    float* outSim  = (float*)d_out + (size_t)B_ * P_ * D_; // [B*P, N]

    const dim3 blk(256);
    k_init_tau<<<1, B_ * PP, 0, stream>>>(tau);
    k_norm_rows<<<B_ * N_, blk, 0, stream>>>(feats, featsN);
    k_norm_rows<<<N_,      blk, 0, stream>>>(featsOrg, forgN);

    // initial sim from input prototypes
    k_norm_protos<<<B_ * PP, blk, 0, stream>>>(protos0, P_, protN);
    k_sim<<<dim3(N_ / 128, B_), blk, 0, stream>>>(protN, featsN, S, 1, 0);

    for (int it = 0; it < 5; ++it) {
        k_softmax<<<dim3(P_, B_), blk, 0, stream>>>(S, tau, Wbuf);
        k_argmax<<<(B_ * N_) / 256, blk, 0, stream>>>(Wbuf, idx);
        k_update<<<dim3(D_ / 128, B_), blk, 0, stream>>>(Wbuf, idx, feats, protoBuf);
        k_norm_protos<<<B_ * PP, blk, 0, stream>>>(protoBuf, PP, protN);
        if (it < 4) {
            // sim2 (density) == next iteration's sim1: compute once, reuse
            k_sim<<<dim3(N_ / 128, B_), blk, 0, stream>>>(protN, featsN, S, 1, 0);
            k_density<<<dim3(P_, B_), blk, 0, stream>>>(S, idx, tau);
        }
    }

    // final sim map vs normalized feats_org, written straight to output
    k_sim<<<dim3(N_ / 128, B_), blk, 0, stream>>>(protN, forgN, outSim, 0, 1);
    k_copy_protos<<<(B_ * P_ * D_) / 256, blk, 0, stream>>>(protoBuf, outProt);
}
